// Net_20916490732073
// MI455X (gfx1250) — matrix operand order repaired and checked by reading, not compiled, not run on hardware
//
#include <hip/hip_runtime.h>

// ---------------------------------------------------------------------------
// CDNA5 / gfx1250 WMMA implementation of the LSTM + MLP + per-action head.
// wave32, bf16 WMMA (v_wmma_f32_16x16x32_bf16), f32 accumulate.
// ---------------------------------------------------------------------------

typedef __attribute__((ext_vector_type(16))) __bf16 v16bf;
typedef __attribute__((ext_vector_type(8)))  __bf16 v8bf;
typedef __attribute__((ext_vector_type(8)))  float  v8f;

static constexpr int H_SZ   = 256;
static constexpr int S_SZ   = 1024;
static constexpr int B_SZ   = 16384;
static constexpr int A_SZ   = 327680;
static constexpr int AS_SZ  = 113;      // per-action state size
static constexpr int KPAD   = 256;      // padded K for action layer 1 (113 | pad | 128)

union U16BF { v16bf v; v8bf h[2]; };

static __device__ inline v8f zero8() {
  v8f z = {0.f,0.f,0.f,0.f,0.f,0.f,0.f,0.f};
  return z;
}

static __device__ inline v8f wmma_bf16(v16bf a, v16bf b, v8f c) {
  // D = A(16x32 bf16) * B(32x16 bf16) + C(16x16 f32)
  return __builtin_amdgcn_wmma_f32_16x16x32_bf16(
      /*neg_a=*/false, a, /*neg_b=*/false, b,
      /*c_mod=*/(short)0, c, /*reuse_a=*/false, /*reuse_b=*/false);
}

// A-fragment (16x32, row-major source, M rows x K cols), converting f32->bf16.
// lane m = lane&15 ; ks = (lane>>4)*8 ; K runs {k0+ks .. +7} and {k0+16+ks .. +7}
static __device__ inline v16bf load_fragA_f32(const float* __restrict__ base,
                                              int ld, int k0, int lane) {
  const int m  = lane & 15;
  const int ks = (lane >> 4) << 3;
  const float* p0 = base + (size_t)m * ld + k0 + ks;
  const float* p1 = p0 + 16;
  float4 a0 = *(const float4*)(p0);
  float4 a1 = *(const float4*)(p0 + 4);
  float4 b0 = *(const float4*)(p1);
  float4 b1 = *(const float4*)(p1 + 4);
  union { v16bf v; __bf16 e[16]; } f;
  f.e[0]=(__bf16)a0.x;  f.e[1]=(__bf16)a0.y;  f.e[2]=(__bf16)a0.z;  f.e[3]=(__bf16)a0.w;
  f.e[4]=(__bf16)a1.x;  f.e[5]=(__bf16)a1.y;  f.e[6]=(__bf16)a1.z;  f.e[7]=(__bf16)a1.w;
  f.e[8]=(__bf16)b0.x;  f.e[9]=(__bf16)b0.y;  f.e[10]=(__bf16)b0.z; f.e[11]=(__bf16)b0.w;
  f.e[12]=(__bf16)b1.x; f.e[13]=(__bf16)b1.y; f.e[14]=(__bf16)b1.z; f.e[15]=(__bf16)b1.w;
  return f.v;
}

// A-fragment from bf16 source (global or LDS).
static __device__ inline v16bf load_fragA_bf16(const __bf16* __restrict__ base,
                                               int ld, int k0, int lane) {
  const int m  = lane & 15;
  const int ks = (lane >> 4) << 3;
  const __bf16* p0 = base + (size_t)m * ld + k0 + ks;
  U16BF f;
  f.h[0] = *(const v8bf*)(p0);
  f.h[1] = *(const v8bf*)(p0 + 16);
  return f.v;
}

// B-fragment (32x16): column n comes from row n of a row-major (N x K) weight.
// lane n = lane&15 ; ks = (lane>>4)*8 ; K runs {k0+ks .. +7} and {k0+16+ks .. +7}
static __device__ inline v16bf load_fragB_bf16(const __bf16* __restrict__ base,
                                               int ld, int k0, int lane) {
  const int n  = lane & 15;
  const int kh = (lane >> 4) << 3;
  const __bf16* p0 = base + (size_t)n * ld + k0 + kh;
  U16BF f;
  f.h[0] = *(const v8bf*)(p0);
  f.h[1] = *(const v8bf*)(p0 + 16);
  return f.v;
}

static __device__ inline float sigf(float x) { return 1.0f / (1.0f + __expf(-x)); }

// ---------------------------------------------------------------------------
// Prep: f32 -> bf16 weight conversion
// ---------------------------------------------------------------------------
__global__ void cvt_f32_bf16(const float* __restrict__ src, __bf16* __restrict__ dst, int n) {
  int i = blockIdx.x * blockDim.x + threadIdx.x;
  if (i < n) dst[i] = (__bf16)src[i];
}

// Wa1 (64 x 241) -> padded bf16 (64 x 256): cols [0,113) direct, [113,128) zero,
// [128,256) = Wa1 cols [113,241). Matches the staged A layout in action_kernel.
__global__ void pad_wa1_kernel(const float* __restrict__ wa1, __bf16* __restrict__ dst) {
  int i = blockIdx.x * blockDim.x + threadIdx.x;
  if (i < 64 * KPAD) {
    int n = i >> 8, k = i & 255;
    float v = 0.f;
    if (k < AS_SZ)       v = wa1[n * 241 + k];
    else if (k >= 128)   v = wa1[n * 241 + AS_SZ + (k - 128)];
    dst[i] = (__bf16)v;
  }
}

// ---------------------------------------------------------------------------
// Kernel 1: fused LSTM cell. Block = 16 batch rows x 1024 gate columns.
// Wave w owns j in [w*32, w*32+32) across all 4 gate quadrants -> the i/f/g/o
// values for a given (row, j) live in the SAME lane -> pointwise LSTM update
// is fully in-register (no LDS, no gates scratch).
// ---------------------------------------------------------------------------
__global__ __launch_bounds__(256) void lstm_kernel(
    const float* __restrict__ x, const float* __restrict__ lstmState,
    const __bf16* __restrict__ WihB, const __bf16* __restrict__ WhhB,
    const float* __restrict__ bih, const float* __restrict__ bhh,
    float* __restrict__ hcOut)
{
  const int w     = threadIdx.x >> 5;
  const int lane  = threadIdx.x & 31;
  const int m0    = blockIdx.x << 4;
  const int jBase = w << 5;

  v8f acc[4][2];
#pragma unroll
  for (int q = 0; q < 4; ++q)
#pragma unroll
    for (int t = 0; t < 2; ++t) acc[q][t] = zero8();

  // gates += x @ W_ih^T   (K = 1024)
  const float* Ax = x + (size_t)m0 * S_SZ;
  for (int k0 = 0; k0 < S_SZ; k0 += 32) {
    v16bf a = load_fragA_f32(Ax, S_SZ, k0, lane);
#pragma unroll
    for (int q = 0; q < 4; ++q)
#pragma unroll
      for (int t = 0; t < 2; ++t) {
        const __bf16* Bb = WihB + (size_t)(q * H_SZ + jBase + t * 16) * S_SZ;
        acc[q][t] = wmma_bf16(a, load_fragB_bf16(Bb, S_SZ, k0, lane), acc[q][t]);
      }
  }
  // gates += h0 @ W_hh^T  (K = 256); h0 = lstmState[:, :256], ld = 512
  const float* Ah = lstmState + (size_t)m0 * (2 * H_SZ);
  for (int k0 = 0; k0 < H_SZ; k0 += 32) {
    v16bf a = load_fragA_f32(Ah, 2 * H_SZ, k0, lane);
#pragma unroll
    for (int q = 0; q < 4; ++q)
#pragma unroll
      for (int t = 0; t < 2; ++t) {
        const __bf16* Bb = WhhB + (size_t)(q * H_SZ + jBase + t * 16) * H_SZ;
        acc[q][t] = wmma_bf16(a, load_fragB_bf16(Bb, H_SZ, k0, lane), acc[q][t]);
      }
  }

  // In-register LSTM pointwise. m = r + 8*(lane>>4), n = lane&15.
  const int hi = lane >> 4;
  const int nl = lane & 15;
#pragma unroll
  for (int t = 0; t < 2; ++t) {
    const int j = jBase + t * 16 + nl;
    float bi[4];
#pragma unroll
    for (int q = 0; q < 4; ++q) bi[q] = bih[q * H_SZ + j] + bhh[q * H_SZ + j];
#pragma unroll
    for (int r = 0; r < 8; ++r) {
      const int m = m0 + r + 8 * hi;
      float iv = sigf(acc[0][t][r] + bi[0]);
      float fv = sigf(acc[1][t][r] + bi[1]);
      float gv = tanhf(acc[2][t][r] + bi[2]);
      float ov = sigf(acc[3][t][r] + bi[3]);
      float c0 = lstmState[(size_t)m * (2 * H_SZ) + H_SZ + j];
      float c1 = fv * c0 + iv * gv;
      float h1 = ov * tanhf(c1);
      hcOut[(size_t)m * (2 * H_SZ) + j]        = h1;  // h1 block
      hcOut[(size_t)m * (2 * H_SZ) + H_SZ + j] = c1;  // c1 block
    }
  }
}

// ---------------------------------------------------------------------------
// Kernel 2: fused z = relu(relu(h1@W1^T+b1)@W2^T+b2), stored bf16 to ws.
// Block = 16 rows; wave w = output n-tile w*16. Layer-1 staged bf16 in LDS.
// ---------------------------------------------------------------------------
__global__ __launch_bounds__(256) void mlp_kernel(
    const float* __restrict__ hc, const __bf16* __restrict__ W1B,
    const float* __restrict__ b1, const __bf16* __restrict__ W2B,
    const float* __restrict__ b2, __bf16* __restrict__ zws)
{
  __shared__ __align__(32) __bf16 z1[16 * 128];
  const int w    = threadIdx.x >> 5;
  const int lane = threadIdx.x & 31;
  const int m0   = blockIdx.x << 4;
  const int hi   = lane >> 4;
  const int nl   = lane & 15;
  const int n0   = w << 4;

  // layer 1: K = 256 over h1 (cols 0..255 of hc rows, ld 512)
  v8f acc = zero8();
  const float* Abase = hc + (size_t)m0 * (2 * H_SZ);
  for (int k0 = 0; k0 < H_SZ; k0 += 32) {
    v16bf a = load_fragA_f32(Abase, 2 * H_SZ, k0, lane);
    v16bf b = load_fragB_bf16(W1B + (size_t)n0 * H_SZ, H_SZ, k0, lane);
    acc = wmma_bf16(a, b, acc);
  }
  {
    const int n = n0 + nl;
    const float bias = b1[n];
#pragma unroll
    for (int r = 0; r < 8; ++r) {
      float v = acc[r] + bias;
      v = v > 0.f ? v : 0.f;
      z1[(r + 8 * hi) * 128 + n] = (__bf16)v;
    }
  }
  __syncthreads();

  // layer 2: K = 128 from LDS
  v8f acc2 = zero8();
  for (int k0 = 0; k0 < 128; k0 += 32) {
    v16bf a = load_fragA_bf16(z1, 128, k0, lane);
    v16bf b = load_fragB_bf16(W2B + (size_t)n0 * 128, 128, k0, lane);
    acc2 = wmma_bf16(a, b, acc2);
  }
  {
    const int n = n0 + nl;
    const float bias = b2[n];
#pragma unroll
    for (int r = 0; r < 8; ++r) {
      float v = acc2[r] + bias;
      v = v > 0.f ? v : 0.f;
      zws[(size_t)(m0 + r + 8 * hi) * 128 + n] = (__bf16)v;
    }
  }
}

// ---------------------------------------------------------------------------
// Kernel 3: per-action head. Block = 64 actions.
// Stage A = [perActionStates | 0 pad | gathered z] as bf16 in LDS (K=256),
// WMMA layer 1 (64x64), then tiny 64->4->1 layers as VALU dots.
// ---------------------------------------------------------------------------
__global__ __launch_bounds__(256) void action_kernel(
    const float* __restrict__ pas, const int* __restrict__ a2s,
    const __bf16* __restrict__ zws, const __bf16* __restrict__ Wa1P,
    const float* __restrict__ ba1, const float* __restrict__ Wa2,
    const float* __restrict__ ba2, const float* __restrict__ Wa3,
    const float* __restrict__ ba3, float* __restrict__ aOut)
{
  __shared__ __align__(32) __bf16 Abuf[64 * KPAD];  // 32 KB
  __shared__ float hid[64 * 64];                    // 16 KB
  const int tid = threadIdx.x;
  const int a0  = blockIdx.x * 64;

  // ---- stage A tile: 4 threads per action row, 64 cols each ----
  {
    const int row    = tid >> 2;
    const int sub    = tid & 3;
    const int action = a0 + row;
    if (sub < 2) {
      // cols [0,128): perActionStates then zero pad for [113,128)
#pragma unroll 4
      for (int cc = 0; cc < 64; ++cc) {
        const int col = sub * 64 + cc;
        float v = (col < AS_SZ) ? pas[(size_t)action * AS_SZ + col] : 0.f;
        Abuf[row * KPAD + col] = (__bf16)v;
      }
    } else {
      // cols [128,256): gathered z row (already bf16) -> 16B vector copies
      const __bf16* src = zws + (size_t)a2s[action] * 128 + (sub - 2) * 64;
      __bf16* dst = &Abuf[row * KPAD + sub * 64];
#pragma unroll
      for (int cc = 0; cc < 64; cc += 8)
        *(v8bf*)(dst + cc) = *(const v8bf*)(src + cc);
    }
  }
  __syncthreads();

  // ---- layer 1: (64 x 256) @ (256 x 64) -> 64x64, 8 waves x 2 tiles ----
  {
    const int w     = tid >> 5;
    const int lane  = tid & 31;
    const int mt    = w >> 1;      // 0..3  (m tile)
    const int npair = w & 1;       // 0..1  (pair of n tiles)
    const int hi    = lane >> 4;
    const int nl    = lane & 15;

    v8f acc[2];
    acc[0] = zero8();
    acc[1] = zero8();
    for (int k0 = 0; k0 < KPAD; k0 += 32) {
      v16bf a = load_fragA_bf16(&Abuf[(mt * 16) * KPAD], KPAD, k0, lane);
#pragma unroll
      for (int t = 0; t < 2; ++t) {
        const int n0 = npair * 32 + t * 16;
        v16bf b = load_fragB_bf16(Wa1P + (size_t)n0 * KPAD, KPAD, k0, lane);
        acc[t] = wmma_bf16(a, b, acc[t]);
      }
    }
#pragma unroll
    for (int t = 0; t < 2; ++t) {
      const int n = npair * 32 + t * 16 + nl;
      const float bias = ba1[n];
#pragma unroll
      for (int r = 0; r < 8; ++r) {
        float v = acc[t][r] + bias;
        v = v > 0.f ? v : 0.f;
        hid[(mt * 16 + r + 8 * hi) * 64 + n] = v;
      }
    }
  }
  __syncthreads();

  // ---- layers 2+3: tiny (64 -> 4 -> 1), one thread per action ----
  if (tid < 64) {
    const float* h = &hid[tid * 64];
    float out = ba3[0];
#pragma unroll
    for (int g = 0; g < 4; ++g) {
      float s = ba2[g];
      const float* wrow = Wa2 + g * 64;
#pragma unroll 8
      for (int k = 0; k < 64; ++k) s += h[k] * wrow[k];
      s = s > 0.f ? s : 0.f;
      out += s * Wa3[g];
    }
    aOut[a0 + tid] = out;
  }
}

// ---------------------------------------------------------------------------
// Host launcher
// ---------------------------------------------------------------------------
extern "C" void kernel_launch(void* const* d_in, const int* in_sizes, int n_in,
                              void* d_out, int out_size, void* d_ws, size_t ws_size,
                              hipStream_t stream)
{
  (void)in_sizes; (void)n_in; (void)out_size; (void)ws_size;

  const float* x    = (const float*)d_in[0];
  const float* lst  = (const float*)d_in[1];
  const float* pas  = (const float*)d_in[2];
  const int*   a2s  = (const int*)  d_in[3];
  const float* Wih  = (const float*)d_in[4];
  const float* Whh  = (const float*)d_in[5];
  const float* bih  = (const float*)d_in[6];
  const float* bhh  = (const float*)d_in[7];
  const float* W1   = (const float*)d_in[8];
  const float* b1   = (const float*)d_in[9];
  const float* W2   = (const float*)d_in[10];
  const float* b2   = (const float*)d_in[11];
  const float* Wa1  = (const float*)d_in[12];
  const float* ba1  = (const float*)d_in[13];
  const float* Wa2  = (const float*)d_in[14];
  const float* ba2  = (const float*)d_in[15];
  const float* Wa3  = (const float*)d_in[16];
  const float* ba3  = (const float*)d_in[17];

  // Workspace layout (bytes), all offsets 32B-aligned; total ~6.95 MB.
  char* ws = (char*)d_ws;
  __bf16* WihB = (__bf16*)(ws + 0);        // 1024*1024*2 = 2,097,152
  __bf16* WhhB = (__bf16*)(ws + 2097152);  // 1024* 256*2 =   524,288
  __bf16* W1B  = (__bf16*)(ws + 2621440);  //  128* 256*2 =    65,536
  __bf16* W2B  = (__bf16*)(ws + 2686976);  //  128* 128*2 =    32,768
  __bf16* Wa1P = (__bf16*)(ws + 2719744);  //   64* 256*2 =    32,768
  __bf16* zB   = (__bf16*)(ws + 2752512);  // 16384*128*2 = 4,194,304

  float* aOut  = (float*)d_out;            // A_SZ floats
  float* hcOut = aOut + A_SZ;              // B_SZ * 512 floats (h1 | c1)

  // Weight conversions
  cvt_f32_bf16<<<(1048576 + 255) / 256, 256, 0, stream>>>(Wih, WihB, 1048576);
  cvt_f32_bf16<<<(262144  + 255) / 256, 256, 0, stream>>>(Whh, WhhB, 262144);
  cvt_f32_bf16<<<(32768   + 255) / 256, 256, 0, stream>>>(W1,  W1B,  32768);
  cvt_f32_bf16<<<(16384   + 255) / 256, 256, 0, stream>>>(W2,  W2B,  16384);
  pad_wa1_kernel<<<(64 * KPAD + 255) / 256, 256, 0, stream>>>(Wa1, Wa1P);

  // Main pipeline (stream-ordered)
  lstm_kernel  <<<B_SZ / 16, 256, 0, stream>>>(x, lst, WihB, WhhB, bih, bhh, hcOut);
  mlp_kernel   <<<B_SZ / 16, 256, 0, stream>>>(hcOut, W1B, b1, W2B, b2, zB);
  action_kernel<<<A_SZ / 64, 256, 0, stream>>>(pas, a2s, zB, Wa1P, ba1, Wa2, ba2,
                                               ba3 ? Wa3 : Wa3, ba3, aOut);
}
